// TransformerBlock_80573586473766
// MI455X (gfx1250) — compile-verified, matrix operand order checked by reading, not run on hardware
//
#include <hip/hip_runtime.h>

typedef __bf16 bf16;
typedef __attribute__((ext_vector_type(16))) __bf16 v16bf;
typedef __attribute__((ext_vector_type(8)))  float  v8f;
typedef int v4i32 __attribute__((vector_size(16)));

#define DEV static __device__ __forceinline__

#if defined(__AMDGCN__) && __has_builtin(__builtin_amdgcn_global_load_async_to_lds_b128)
#define USE_ASYNC 1
#else
#define USE_ASYNC 0
#endif

DEV float bf2f(bf16 x) { return (float)x; }
DEV bf16  f2bf(float x) { return (bf16)x; }

DEV float waveSum(float v) {
#pragma unroll
  for (int off = 16; off > 0; off >>= 1) v += __shfl_xor(v, off, 32);
  return v;
}

template <int N> DEV void waitAsync() {
#if defined(__AMDGCN__)
#if __has_builtin(__builtin_amdgcn_s_wait_asynccnt)
  __builtin_amdgcn_s_wait_asynccnt((short)N);
#else
  asm volatile("s_wait_asynccnt %0" ::"i"(N));
#endif
#endif
}

// One 16-byte async copy global -> LDS (per-lane addresses), ASYNCcnt-tracked.
DEV void asyncLoad16(const bf16* g, bf16* l) {
#if USE_ASYNC
  __builtin_amdgcn_global_load_async_to_lds_b128(
      (__attribute__((address_space(1))) v4i32*)(void*)g,
      (__attribute__((address_space(3))) v4i32*)(void*)l, 0, 0);
#else
  (void)g; (void)l;
#endif
}

// ---------------------------------------------------------------------------
// Generic batched NT GEMM:  C[p] = A[p] (MxK row-major) * Bt[p] (NxK row-major)^T
// bf16 inputs, f32 accumulation via v_wmma_f32_16x16x32_bf16 (wave32).
// Block = 256 threads (8 waves). CTA tile 64x128, wave tile 16x64, K-step 32.
// Double-buffered LDS staging via GLOBAL_LOAD_ASYNC_TO_LDS_B128 when available.
// Per-operand batch offset: off(p) = (p/bdiv)*Out + (p%bdiv)*In.
// C addressing: addr = cOff + row*ldcRow + col*ldcCol.
// TS=true: transposed store (ldcRow==1), 8 consecutive bf16 packed per 16B store.
// Epilogues (EPI):
//   0: f32  out = acc + bias[col]
//   1: bf16 out = (acc + bias[col]) * scale
//   2: bf16 out = relu(acc + bias[col])
//   3: bf16 out = exp(acc - aux[p*auxStride+row]) * scale + 1e-6   (FAVOR+ phi)
//   4: bf16 out = acc * aux[p*auxStride+row]                        (1/z rescale)
// Requires: M % 64 == 0, N % 128 == 0, K % 32 == 0 (true for all uses here).
// ---------------------------------------------------------------------------
template <int EPI, bool TS>
__global__ __launch_bounds__(256) void gemm_nt(
    const bf16* __restrict__ A, const bf16* __restrict__ Bt, void* __restrict__ Cv,
    const float* __restrict__ bias, const float* __restrict__ aux,
    int K, int lda, int ldb,
    long long ldcRow, long long ldcCol, int bdiv,
    long long aOut, long long aIn,
    long long bOut, long long bIn,
    long long cOut, long long cIn,
    long long auxStride, float scale)
{
  constexpr int SA = 64 * 48;   // one A buffer: 64 rows x 32 k, padded stride 48
  constexpr int SB = 128 * 48;  // one B buffer: 128 rows x 32 k
  __shared__ bf16 sA[2 * SA];
  __shared__ bf16 sB[2 * SB];

  const int t    = threadIdx.x;
  const int lane = t & 31;
  const int wave = t >> 5;
  const int wm   = wave & 3;   // wave M slot (x16)
  const int wn   = wave >> 2;  // wave N slot (x64)

  const int p = blockIdx.z;
  const long long po = p / bdiv, pi = p % bdiv;
  A  += po * aOut + pi * aIn;
  Bt += po * bOut + pi * bIn;
  const long long cOff = po * cOut + pi * cIn;

  const int rowBase = blockIdx.x * 64;
  const int colBase = blockIdx.y * 128;

  // staging map: A -> 64 rows x 4 chunks(8 bf16); B -> 128 rows x 4 chunks (2/thread)
  const int ar = t >> 2, ac = (t & 3) * 8;

  v8f acc[4] = {};

  auto computeTile = [&](int buf) {
    const bf16* a0 = &sA[buf * SA];
    const bf16* b0 = &sB[buf * SB];
    union { uint4 u[2]; v16bf v; } fa;
    const int aoff = (wm * 16 + (lane & 15)) * 48 + (lane >> 4) * 8;
    fa.u[0] = *(const uint4*)(&a0[aoff]);
    fa.u[1] = *(const uint4*)(&a0[aoff + 16]);
#pragma unroll
    for (int f = 0; f < 4; ++f) {
      union { uint4 u[2]; v16bf v; } fb;
      const int boff = (wn * 64 + f * 16 + (lane & 15)) * 48 + (lane >> 4) * 8;
      fb.u[0] = *(const uint4*)(&b0[boff]);
      fb.u[1] = *(const uint4*)(&b0[boff + 16]);
      acc[f] = __builtin_amdgcn_wmma_f32_16x16x32_bf16(
          false, fa.v, false, fb.v, (short)0, acc[f], false, false);
    }
  };

#if USE_ASYNC
  auto issueStage = [&](int kk, int buf) {
    const bf16* ga = A + (long long)(rowBase + ar) * lda + kk + ac;
    asyncLoad16(ga, &sA[buf * SA + ar * 48 + ac]);
#pragma unroll
    for (int i = 0; i < 2; ++i) {
      const int slot = t + i * 256;
      const int brow = slot >> 2, bc = (slot & 3) * 8;
      const bf16* gb = Bt + (long long)(colBase + brow) * ldb + kk + bc;
      asyncLoad16(gb, &sB[buf * SB + brow * 48 + bc]);
    }
  };
  issueStage(0, 0);
  int ibuf = 0;
  for (int kk = 0; kk < K; kk += 32, ibuf ^= 1) {
    const bool more = (kk + 32) < K;
    if (more) {
      issueStage(kk + 32, ibuf ^ 1);  // prefetch next stage into other buffer
      waitAsync<3>();                 // current stage (3 older copies) landed
    } else {
      waitAsync<0>();
    }
    __syncthreads();     // all waves' current stage visible
    computeTile(ibuf);
    __syncthreads();     // done reading before buffer is overwritten next iter
  }
#else
  for (int kk = 0; kk < K; kk += 32) {
    {
      const bf16* ga = A + (long long)(rowBase + ar) * lda + kk + ac;
      *(uint4*)(&sA[ar * 48 + ac]) = *(const uint4*)ga;
    }
#pragma unroll
    for (int i = 0; i < 2; ++i) {
      const int slot = t + i * 256;
      const int brow = slot >> 2, bc = (slot & 3) * 8;
      const bf16* gb = Bt + (long long)(colBase + brow) * ldb + kk + bc;
      *(uint4*)(&sB[brow * 48 + bc]) = *(const uint4*)gb;
    }
    __syncthreads();
    computeTile(0);
    __syncthreads();
  }
#endif

  float* Cf = (float*)Cv;
  bf16*  Cb = (bf16*)Cv;
  if (TS) {
    // transposed store: ldcRow==1, 8 consecutive rows per lane -> one 16B store
#pragma unroll
    for (int f = 0; f < 4; ++f) {
      const int col = colBase + wn * 64 + f * 16 + (lane & 15);
      const float bc = bias ? bias[col] : 0.0f;
      const int row0 = rowBase + wm * 16 + ((lane >> 4) << 3);
      union { uint4 u; bf16 h[8]; } pk;
#pragma unroll
      for (int r = 0; r < 8; ++r) {
        const float v = acc[f][r];
        float o;
        if (EPI == 1)      o = (v + bc) * scale;
        else if (EPI == 2) { o = v + bc; o = o > 0.0f ? o : 0.0f; }
        else if (EPI == 3) o = __expf(v - aux[(long long)p * auxStride + row0 + r]) * scale + 1e-6f;
        else if (EPI == 4) o = v * aux[(long long)p * auxStride + row0 + r];
        else               o = v + bc;
        pk.h[r] = f2bf(o);
      }
      *(uint4*)(&Cb[cOff + (long long)col * ldcCol + row0]) = pk.u;
    }
  } else {
#pragma unroll
    for (int f = 0; f < 4; ++f) {
      const int col = colBase + wn * 64 + f * 16 + (lane & 15);
      const float bc = bias ? bias[col] : 0.0f;
#pragma unroll
      for (int r = 0; r < 8; ++r) {
        const int row = rowBase + wm * 16 + ((lane >> 4) << 3) + r;
        const float v = acc[f][r];
        const long long addr = cOff + (long long)row * ldcRow + (long long)col * ldcCol;
        if (EPI == 0) {
          Cf[addr] = v + bc;
        } else if (EPI == 1) {
          Cb[addr] = f2bf((v + bc) * scale);
        } else if (EPI == 2) {
          const float o = v + bc;
          Cb[addr] = f2bf(o > 0.0f ? o : 0.0f);
        } else if (EPI == 3) {
          const float sq = aux[(long long)p * auxStride + row];
          Cb[addr] = f2bf(__expf(v - sq) * scale + 1e-6f);
        } else {
          const float iz = aux[(long long)p * auxStride + row];
          Cb[addr] = f2bf(v * iz);
        }
      }
    }
  }
}

// ---------------------------------------------------------------------------
// Small helper kernels
// ---------------------------------------------------------------------------

// out[c*R + r] = bf16(in[r*C + c])   (weight transpose to N-by-K "Bt" layout)
__global__ __launch_bounds__(256) void transpose_to_bf16(
    const float* __restrict__ in, bf16* __restrict__ out, int R, int C)
{
  const long long i = (long long)blockIdx.x * 256 + threadIdx.x;
  if (i < (long long)R * C) {
    const int r = (int)(i / C), c = (int)(i % C);
    out[(long long)c * R + r] = f2bf(in[i]);
  }
}

__global__ __launch_bounds__(256) void convert_to_bf16(
    const float* __restrict__ in, bf16* __restrict__ out, long long n)
{
  const long long i = (long long)blockIdx.x * 256 + threadIdx.x;
  if (i < n) out[i] = f2bf(in[i]);
}

// sq[(b*H+h)*S + s] = 0.5 * sum_kd q[b,s,h,kd]^2    (wave per row, 8 rows/block)
__global__ __launch_bounds__(256) void compute_sq(
    const bf16* __restrict__ q, float* __restrict__ sq, int S, int H, int KD)
{
  const int idx  = blockIdx.x * 8 + (threadIdx.x >> 5);  // over B*H*S
  const int lane = threadIdx.x & 31;
  const int p = idx / S, s = idx % S;
  const int b = p / H, h = p % H;
  const bf16* ptr = q + ((long long)(b * S + s) * H + h) * KD;
  float acc = 0.0f;
#pragma unroll
  for (int i = 0; i < 8; ++i) {
    const float v = bf2f(ptr[lane + i * 32]);
    acc += v * v;
  }
  acc = waveSum(acc);
  if (lane == 0) sq[idx] = 0.5f * acc;
}

// ksum[row] = sum_s phi_kT[row*S + s]    rows = B*H*M
__global__ __launch_bounds__(256) void compute_ksum(
    const bf16* __restrict__ phikT, float* __restrict__ ksum, int S)
{
  const int row  = blockIdx.x * 8 + (threadIdx.x >> 5);
  const int lane = threadIdx.x & 31;
  const bf16* ptr = phikT + (long long)row * S;
  float acc = 0.0f;
  for (int j = lane; j < S; j += 32) acc += bf2f(ptr[j]);
  acc = waveSum(acc);
  if (lane == 0) ksum[row] = acc;
}

// invz[p*S + s] = 1 / sum_m phi_q2[p,s,m] * ksum[p,m]
__global__ __launch_bounds__(256) void compute_invz(
    const bf16* __restrict__ phiq2, const float* __restrict__ ksum,
    float* __restrict__ invz, int S, int M)
{
  const int idx  = blockIdx.x * 8 + (threadIdx.x >> 5);  // over B*H*S
  const int lane = threadIdx.x & 31;
  const int p = idx / S, s = idx % S;
  const bf16*  ptr = phiq2 + ((long long)p * S + s) * M;
  const float* kb  = ksum + (long long)p * M;
  float acc = 0.0f;
#pragma unroll
  for (int i = 0; i < 8; ++i) {
    const int j = lane + i * 32;
    acc += bf2f(ptr[j]) * kb[j];
  }
  acc = waveSum(acc);
  if (lane == 0) invz[idx] = 1.0f / acc;
}

// out = LayerNorm(X + Y) * g + beta ; optionally also bf16 copy. Wave/row, D=256.
template <bool WB>
__global__ __launch_bounds__(256) void residual_ln(
    const float* __restrict__ X, const float* __restrict__ Y,
    const float* __restrict__ g, const float* __restrict__ beta,
    float* __restrict__ out, bf16* __restrict__ outb)
{
  const int row  = blockIdx.x * 8 + (threadIdx.x >> 5);
  const int lane = threadIdx.x & 31;
  const long long base = (long long)row * 256;
  float v[8];
  float s = 0.0f;
#pragma unroll
  for (int i = 0; i < 8; ++i) {
    const int j = lane + i * 32;
    v[i] = X[base + j] + Y[base + j];
    s += v[i];
  }
  s = waveSum(s);
  const float mu = s * (1.0f / 256.0f);
  float q = 0.0f;
#pragma unroll
  for (int i = 0; i < 8; ++i) {
    const float d = v[i] - mu;
    q += d * d;
  }
  q = waveSum(q);
  const float rstd = rsqrtf(q * (1.0f / 256.0f) + 1e-6f);
#pragma unroll
  for (int i = 0; i < 8; ++i) {
    const int j = lane + i * 32;
    const float o = (v[i] - mu) * rstd * g[j] + beta[j];
    out[base + j] = o;
    if (WB) outb[base + j] = f2bf(o);
  }
}

// ---------------------------------------------------------------------------
// Orchestration
// ---------------------------------------------------------------------------
extern "C" void kernel_launch(void* const* d_in, const int* in_sizes, int n_in,
                              void* d_out, int out_size, void* d_ws, size_t ws_size,
                              hipStream_t stream)
{
  (void)in_sizes; (void)n_in; (void)out_size; (void)ws_size;

  const float* x    = (const float*)d_in[0];
  const float* Wq   = (const float*)d_in[1];
  const float* bq   = (const float*)d_in[2];
  const float* Wk   = (const float*)d_in[3];
  const float* bk   = (const float*)d_in[4];
  const float* Wv   = (const float*)d_in[5];
  const float* bv   = (const float*)d_in[6];
  const float* Wo   = (const float*)d_in[7];
  const float* bo   = (const float*)d_in[8];
  const float* proj = (const float*)d_in[9];
  const float* W1   = (const float*)d_in[10];
  const float* b1   = (const float*)d_in[11];
  const float* W2   = (const float*)d_in[12];
  const float* b2   = (const float*)d_in[13];
  const float* g1   = (const float*)d_in[14];
  const float* be1  = (const float*)d_in[15];
  const float* g2   = (const float*)d_in[16];
  const float* be2  = (const float*)d_in[17];
  float* outp = (float*)d_out;

  constexpr int  Bb = 8, S = 2048, D = 256, H = 8, KD = 256, Mf = 256;
  constexpr long long NT = (long long)Bb * S;   // 16384 tokens
  constexpr long long NR = NT * H;              // 131072 rows (b,h,s)

  char* ws = (char*)d_ws;
  size_t cur = 0;
  auto alloc = [&](size_t bytes) -> char* {
    char* ptr = ws + cur;
    cur += (bytes + 255) & ~size_t(255);
    return ptr;
  };

  bf16* x_bf    = (bf16*)alloc(NT * D * 2);
  bf16* Wqt     = (bf16*)alloc((size_t)D * H * KD * 2);
  bf16* Wkt     = (bf16*)alloc((size_t)D * H * KD * 2);
  bf16* Wvt     = (bf16*)alloc((size_t)D * H * KD * 2);
  bf16* Wot     = (bf16*)alloc((size_t)H * KD * D * 2);
  bf16* W1t     = (bf16*)alloc((size_t)D * D * 2);
  bf16* W2t     = (bf16*)alloc((size_t)D * D * 2);
  bf16* proj_bf = (bf16*)alloc((size_t)Mf * KD * 2);
  bf16* q_bf    = (bf16*)alloc((size_t)NT * H * KD * 2);  // [b,s,h,kd]; reused as attn
  bf16* k_bf    = (bf16*)alloc((size_t)NT * H * KD * 2);  // [b,s,h,kd]
  bf16* vT_bf   = (bf16*)alloc((size_t)NT * H * KD * 2);  // [b,h,kd,s]
  bf16* phi_q2  = (bf16*)alloc((size_t)NR * Mf * 2);      // [b,h,s,m]
  bf16* phi_kT  = (bf16*)alloc((size_t)NR * Mf * 2);      // [b,h,m,s]
  float* sqQ    = (float*)alloc(NR * 4);                  // [b,h,s]
  float* sqK    = (float*)alloc(NR * 4);
  float* ksum   = (float*)alloc((size_t)Bb * H * Mf * 4); // [b,h,m]
  float* invZ   = (float*)alloc(NR * 4);                  // [b,h,s]
  bf16* kvT_bf  = (bf16*)alloc((size_t)Bb * H * KD * Mf * 2); // [b,h,kd,m]
  float* attn_o = (float*)alloc(NT * D * 4);
  float* out1   = (float*)alloc(NT * D * 4);
  bf16* out1_bf = (bf16*)alloc(NT * D * 2);
  bf16* attn_bf = q_bf;    // q dead after phi_q / sqQ
  bf16* h_bf    = x_bf;    // x_bf dead after QKV projections
  float* ffn    = attn_o;  // attn_out dead after LN1

  const dim3 blk(256);

  // ---- weight / input conversion ----
  transpose_to_bf16<<<dim3((D * H * KD + 255) / 256), blk, 0, stream>>>(Wq, Wqt, D, H * KD);
  transpose_to_bf16<<<dim3((D * H * KD + 255) / 256), blk, 0, stream>>>(Wk, Wkt, D, H * KD);
  transpose_to_bf16<<<dim3((D * H * KD + 255) / 256), blk, 0, stream>>>(Wv, Wvt, D, H * KD);
  transpose_to_bf16<<<dim3((H * KD * D + 255) / 256), blk, 0, stream>>>(Wo, Wot, H * KD, D);
  transpose_to_bf16<<<dim3((D * D + 255) / 256), blk, 0, stream>>>(W1, W1t, D, D);
  transpose_to_bf16<<<dim3((D * D + 255) / 256), blk, 0, stream>>>(W2, W2t, D, D);
  convert_to_bf16<<<dim3((int)((Mf * KD + 255) / 256)), blk, 0, stream>>>(proj, proj_bf, (long long)Mf * KD);
  convert_to_bf16<<<dim3((int)((NT * D + 255) / 256)), blk, 0, stream>>>(x, x_bf, NT * D);

  // ---- QKV projections (batched over b): q,k scaled by KD^-0.25 = 0.25 ----
  gemm_nt<1, false><<<dim3(S / 64, (H * KD) / 128, Bb), blk, 0, stream>>>(
      x_bf, Wqt, q_bf, bq, nullptr, D, D, D,
      (long long)(H * KD), 1, 1,
      (long long)S * D, 0, 0, 0, (long long)S * H * KD, 0, 0, 0.25f);
  gemm_nt<1, false><<<dim3(S / 64, (H * KD) / 128, Bb), blk, 0, stream>>>(
      x_bf, Wkt, k_bf, bk, nullptr, D, D, D,
      (long long)(H * KD), 1, 1,
      (long long)S * D, 0, 0, 0, (long long)S * H * KD, 0, 0, 0.25f);
  // v with transposed store -> vT[b,h,kd,s]
  gemm_nt<1, true><<<dim3(S / 64, (H * KD) / 128, Bb), blk, 0, stream>>>(
      x_bf, Wvt, vT_bf, bv, nullptr, D, D, D,
      1, (long long)S, 1,
      (long long)S * D, 0, 0, 0, (long long)H * KD * S, 0, 0, 1.0f);

  // ---- 0.5*||q||^2, 0.5*||k||^2 ----
  compute_sq<<<dim3((int)(NR / 8)), blk, 0, stream>>>(q_bf, sqQ, S, H, KD);
  compute_sq<<<dim3((int)(NR / 8)), blk, 0, stream>>>(k_bf, sqK, S, H, KD);

  // ---- phi(q) -> [b,h,s,m] ; phi(k) -> transposed [b,h,m,s] ; scale=1/sqrt(M)=1/16 ----
  gemm_nt<3, false><<<dim3(S / 64, Mf / 128, Bb * H), blk, 0, stream>>>(
      q_bf, proj_bf, phi_q2, nullptr, sqQ, KD, H * KD, KD,
      (long long)Mf, 1, H,
      (long long)S * H * KD, KD, 0, 0,
      (long long)H * S * Mf, (long long)S * Mf, S, 0.0625f);
  gemm_nt<3, true><<<dim3(S / 64, Mf / 128, Bb * H), blk, 0, stream>>>(
      k_bf, proj_bf, phi_kT, nullptr, sqK, KD, H * KD, KD,
      1, (long long)S, H,
      (long long)S * H * KD, KD, 0, 0,
      (long long)H * Mf * S, (long long)Mf * S, S, 0.0625f);

  // ---- ksum[b,h,m] and invZ[b,h,s] ----
  compute_ksum<<<dim3(Bb * H * Mf / 8), blk, 0, stream>>>(phi_kT, ksum, S);
  compute_invz<<<dim3((int)(NR / 8)), blk, 0, stream>>>(phi_q2, ksum, invZ, S, Mf);

  // ---- kvT[b,h,kd,m] = vT[b,h] (KDxS) x phi_kT[b,h] (MxS)^T ----
  gemm_nt<1, false><<<dim3(KD / 64, Mf / 128, Bb * H), blk, 0, stream>>>(
      vT_bf, phi_kT, kvT_bf, nullptr, nullptr, S, S, S,
      (long long)Mf, 1, 1,
      (long long)KD * S, 0, (long long)Mf * S, 0,
      (long long)KD * Mf, 0, 0, 1.0f);

  // ---- attn[b,s,h,kd] = (phi_q2[b,h] (SxM) x kvT[b,h] (KDxM)^T) * invZ ----
  gemm_nt<4, false><<<dim3(S / 64, KD / 128, Bb * H), blk, 0, stream>>>(
      phi_q2, kvT_bf, attn_bf, nullptr, invZ, Mf, Mf, Mf,
      (long long)(H * KD), 1, H,
      (long long)H * S * Mf, (long long)S * Mf,
      (long long)H * KD * Mf, (long long)KD * Mf,
      (long long)S * H * KD, KD, S, 1.0f);

  // ---- attn_out = attn [16384 x 2048] x Wo + bo (f32) ----
  gemm_nt<0, false><<<dim3((int)(NT / 64), D / 128, 1), blk, 0, stream>>>(
      attn_bf, Wot, attn_o, bo, nullptr, H * KD, H * KD, H * KD,
      (long long)D, 1, 1, 0, 0, 0, 0, 0, 0, 0, 1.0f);

  // ---- out1 = LN(x + attn_out) (f32 + bf16 copy) ----
  residual_ln<true><<<dim3((int)(NT / 8)), blk, 0, stream>>>(
      x, attn_o, g1, be1, out1, out1_bf);

  // ---- FFN ----
  gemm_nt<2, false><<<dim3((int)(NT / 64), D / 128, 1), blk, 0, stream>>>(
      out1_bf, W1t, h_bf, b1, nullptr, D, D, D,
      (long long)D, 1, 1, 0, 0, 0, 0, 0, 0, 0, 1.0f);
  gemm_nt<0, false><<<dim3((int)(NT / 64), D / 128, 1), blk, 0, stream>>>(
      h_bf, W2t, ffn, b2, nullptr, D, D, D,
      (long long)D, 1, 1, 0, 0, 0, 0, 0, 0, 0, 1.0f);

  // ---- out = LN(out1 + ffn) ----
  residual_ln<false><<<dim3((int)(NT / 8)), blk, 0, stream>>>(
      out1, ffn, g2, be2, outp, nullptr);
}
